// RazorMambaLSTMCell_38371237822916
// MI455X (gfx1250) — matrix operand order repaired and checked by reading, not compiled, not run on hardware
//
#include <hip/hip_runtime.h>

// ---- problem constants ----
#define B_ROWS 4096
#define IN_DIM 2048
#define H_DIM  2048
#define C_DIM  4096            // IN + H (K of the big GEMM)
#define N_TOT  14464           // 4*2048 gates + 6272 in_proj columns
#define BH     ((size_t)B_ROWS * H_DIM)

// ---- vector types ----
typedef __attribute__((ext_vector_type(16))) __bf16        v16bf;
typedef __attribute__((ext_vector_type(8)))  float         v8f;
typedef __attribute__((ext_vector_type(4)))  unsigned int  u32x4;
typedef __attribute__((ext_vector_type(8)))  unsigned int  u32x8;
typedef __attribute__((ext_vector_type(4)))  float         f32x4;
typedef __attribute__((ext_vector_type(4)))  int           v4i;

// ---- CDNA5 async-load-to-LDS availability (guarded; falls back to reg staging) ----
#if defined(__AMDGCN__) && __has_builtin(__builtin_amdgcn_global_load_async_to_lds_b128)
#define HAVE_ASYNC_LDS 1
#else
#define HAVE_ASYNC_LDS 0
#endif

#if HAVE_ASYNC_LDS
typedef __attribute__((address_space(1))) v4i as1_v4i;
typedef __attribute__((address_space(3))) v4i as3_v4i;
#define ASYNC_CP_B128(gp, lp) \
  __builtin_amdgcn_global_load_async_to_lds_b128((as1_v4i*)(gp), (as3_v4i*)(lp), 0, 0)
#if __has_builtin(__builtin_amdgcn_s_wait_asynccnt)
#define WAIT_ASYNC(n) __builtin_amdgcn_s_wait_asynccnt(n)
#else
#define WAIT_ASYNC(n) asm volatile("s_wait_asynccnt %0" :: "i"(n) : "memory")
#endif
#endif

static __device__ __forceinline__ unsigned short f2bf(float f) {
  unsigned int u = __float_as_uint(f);
  u += 0x7FFFu + ((u >> 16) & 1u);          // round-to-nearest-even
  return (unsigned short)(u >> 16);
}
static __device__ __forceinline__ float sigm(float x) { return 1.0f / (1.0f + expf(-x)); }

// ============================================================
// 1) combined = concat([h, x], axis=1), f32 -> bf16
// ============================================================
__global__ __launch_bounds__(256) void pack_combined(const float* __restrict__ h,
                                                     const float* __restrict__ x,
                                                     unsigned short* __restrict__ cb) {
  int g = blockIdx.x * 256 + threadIdx.x;   // each thread: 4 elements
  int base = g * 4;
  int b = base >> 12;                       // / C_DIM
  int k = base & (C_DIM - 1);
  const float* src = (k < H_DIM) ? (h + (size_t)b * H_DIM + k)
                                 : (x + (size_t)b * IN_DIM + (k - H_DIM));
  f32x4 v = *(const f32x4*)src;
  unsigned int p0 = (unsigned int)f2bf(v.x) | ((unsigned int)f2bf(v.y) << 16);
  unsigned int p1 = (unsigned int)f2bf(v.z) | ((unsigned int)f2bf(v.w) << 16);
  unsigned int* d = (unsigned int*)(cb + base);
  d[0] = p0; d[1] = p1;
}

// ============================================================
// 2) WT[n][k] = bf16(W[k][n]) for concatenated [Wi|Wf|Wo|Wg|in_proj]
// ============================================================
__global__ void wt_build(const float* __restrict__ Wi, const float* __restrict__ Wf,
                         const float* __restrict__ Wo, const float* __restrict__ Wg,
                         const float* __restrict__ Wp, unsigned short* __restrict__ WT) {
  __shared__ float t[32][33];
  const int kt = blockIdx.x * 32;
  const int nt = blockIdx.y * 32;
  const float* src; int width, coff;
  if      (nt < 2048) { src = Wi; width = 2048; coff = nt; }
  else if (nt < 4096) { src = Wf; width = 2048; coff = nt - 2048; }
  else if (nt < 6144) { src = Wo; width = 2048; coff = nt - 4096; }
  else if (nt < 8192) { src = Wg; width = 2048; coff = nt - 6144; }
  else                { src = Wp; width = 6272; coff = nt - 8192; }
  const int tx = threadIdx.x, ty = threadIdx.y;
#pragma unroll
  for (int i = 0; i < 4; ++i) {
    int r = ty + i * 8;
    t[r][tx] = src[(size_t)(kt + r) * width + coff + tx];
  }
  __syncthreads();
#pragma unroll
  for (int i = 0; i < 4; ++i) {
    int n = ty + i * 8;
    WT[(size_t)(nt + n) * C_DIM + kt + tx] = f2bf(t[tx][n]);
  }
}

// ============================================================
// 3) bf16 WMMA GEMM: D[4096][14464] = combined(bf16) @ W(bf16), f32 acc
//    Block 128x128, 8 waves (2x4), wave = 64x32 (4x2 tiles), BK = 32.
//    Double-buffered LDS fed by GLOBAL_LOAD_ASYNC_TO_LDS_B128 (ASYNCcnt),
//    falling back to register staging if the builtin is unavailable.
// ============================================================
__global__ __launch_bounds__(256) void gemm_bf16(const unsigned short* __restrict__ A,
                                                 const unsigned short* __restrict__ Bw,
                                                 float* __restrict__ Dm) {
  __shared__ __align__(16) unsigned short As[2][128 * 40];  // stride 40: conflict-free b128
  __shared__ __align__(16) unsigned short Bs[2][128 * 40];
  const int tid  = threadIdx.x;
  const int m0   = blockIdx.y * 128;
  const int n0   = blockIdx.x * 128;
  const int lrow = tid >> 2;       // 0..63
  const int lci  = tid & 3;        // 16B chunk within 64B row

  const unsigned short* ga = A  + (size_t)(m0 + lrow) * C_DIM + lci * 8;
  const unsigned short* gb = Bw + (size_t)(n0 + lrow) * C_DIM + lci * 8;

  const int lane = tid & 31;
  const int wave = tid >> 5;
  const int wm   = wave & 1;       // 2 waves along M (64 each)
  const int wn   = wave >> 1;      // 4 waves along N (32 each)
  const int lm   = lane & 15;
  const int lh   = lane >> 4;

  const v8f vzero = {0.f, 0.f, 0.f, 0.f, 0.f, 0.f, 0.f, 0.f};
  v8f acc[4][2];
#pragma unroll
  for (int mt = 0; mt < 4; ++mt)
#pragma unroll
    for (int nt = 0; nt < 2; ++nt) acc[mt][nt] = vzero;

  const int KSTEPS = C_DIM / 32;

#if HAVE_ASYNC_LDS
  {   // stage 0 -> buffer 0 (DMA directly into LDS, no VGPR staging)
    unsigned short* la = &As[0][lrow * 40 + lci * 8];
    unsigned short* lb = &Bs[0][lrow * 40 + lci * 8];
    ASYNC_CP_B128(ga,                      la);
    ASYNC_CP_B128(ga + (size_t)64 * C_DIM, la + 64 * 40);
    ASYNC_CP_B128(gb,                      lb);
    ASYNC_CP_B128(gb + (size_t)64 * C_DIM, lb + 64 * 40);
  }
#else
  u32x4 ra0 = *(const u32x4*)ga;
  u32x4 ra1 = *(const u32x4*)(ga + (size_t)64 * C_DIM);
  u32x4 rb0 = *(const u32x4*)gb;
  u32x4 rb1 = *(const u32x4*)(gb + (size_t)64 * C_DIM);
#endif

  for (int ks = 0; ks < KSTEPS; ++ks) {
#if HAVE_ASYNC_LDS
    const int buf = ks & 1;
    if (ks + 1 < KSTEPS) {       // issue next stage into the other buffer
      const unsigned short* pa = ga + (size_t)(ks + 1) * 32;
      const unsigned short* pb = gb + (size_t)(ks + 1) * 32;
      unsigned short* la = &As[buf ^ 1][lrow * 40 + lci * 8];
      unsigned short* lb = &Bs[buf ^ 1][lrow * 40 + lci * 8];
      ASYNC_CP_B128(pa,                      la);
      ASYNC_CP_B128(pa + (size_t)64 * C_DIM, la + 64 * 40);
      ASYNC_CP_B128(pb,                      lb);
      ASYNC_CP_B128(pb + (size_t)64 * C_DIM, lb + 64 * 40);
      WAIT_ASYNC(4);             // in-order: retires current stage's 4 loads
    } else {
      WAIT_ASYNC(0);
    }
    __syncthreads();             // all waves' stage data resident in LDS
#else
    const int buf = 0;
    *(u32x4*)(&As[0][lrow * 40 + lci * 8])        = ra0;
    *(u32x4*)(&As[0][(lrow + 64) * 40 + lci * 8]) = ra1;
    *(u32x4*)(&Bs[0][lrow * 40 + lci * 8])        = rb0;
    *(u32x4*)(&Bs[0][(lrow + 64) * 40 + lci * 8]) = rb1;
    __syncthreads();
    if (ks + 1 < KSTEPS) {
      const unsigned short* pa = ga + (size_t)(ks + 1) * 32;
      const unsigned short* pb = gb + (size_t)(ks + 1) * 32;
      ra0 = *(const u32x4*)pa; ra1 = *(const u32x4*)(pa + (size_t)64 * C_DIM);
      rb0 = *(const u32x4*)pb; rb1 = *(const u32x4*)(pb + (size_t)64 * C_DIM);
    }
#endif
    if (ks + 8 < KSTEPS) {       // L2 prefetch ahead -> global_prefetch_b8
      __builtin_prefetch(ga + (size_t)(ks + 8) * 32, 0, 1);
      __builtin_prefetch(gb + (size_t)(ks + 8) * 32, 0, 1);
    }

    // A fragments: 16x32 bf16, lane layout per ISA 7.12.2
    v16bf af[4];
#pragma unroll
    for (int mt = 0; mt < 4; ++mt) {
      const unsigned short* p = &As[buf][(wm * 64 + mt * 16 + lm) * 40 + lh * 8];
      u32x4 lo = *(const u32x4*)p;          // K = kb .. kb+7
      u32x4 hi = *(const u32x4*)(p + 16);   // K = 16+kb .. 16+kb+7
      u32x8 w = __builtin_shufflevector(lo, hi, 0, 1, 2, 3, 4, 5, 6, 7);
      af[mt] = __builtin_bit_cast(v16bf, w);
    }
    // B fragments from transposed tile: lane holds K = 8*lh..+7 and 16+8*lh..+7 of column n
#pragma unroll
    for (int nt = 0; nt < 2; ++nt) {
      const unsigned short* p = &Bs[buf][(wn * 32 + nt * 16 + lm) * 40 + lh * 8];
      u32x4 lo = *(const u32x4*)p;
      u32x4 hi = *(const u32x4*)(p + 16);
      u32x8 w = __builtin_shufflevector(lo, hi, 0, 1, 2, 3, 4, 5, 6, 7);
      v16bf bfr = __builtin_bit_cast(v16bf, w);
#pragma unroll
      for (int mt = 0; mt < 4; ++mt) {
        acc[mt][nt] = __builtin_amdgcn_wmma_f32_16x16x32_bf16(
            false, af[mt], false, bfr, (short)0, acc[mt][nt], false, false);
      }
    }
    __syncthreads();             // protect buffer being refilled next iteration
  }

  // epilogue: C/D layout -> M = r + 8*lh, N = lm
  float* dout = Dm + (size_t)(m0 + wm * 64 + 8 * lh) * N_TOT + n0 + wn * 32 + lm;
#pragma unroll
  for (int mt = 0; mt < 4; ++mt)
#pragma unroll
    for (int nt = 0; nt < 2; ++nt)
#pragma unroll
      for (int r = 0; r < 8; ++r)
        dout[(size_t)(mt * 16 + r) * N_TOT + nt * 16] = acc[mt][nt][r];
}

// ============================================================
// 4) dt = softplus(dt_r @ dt_proj_w + dt_proj_b)  (K=128, f32 FMA)
// ============================================================
__global__ __launch_bounds__(256) void dt_gemm(const float* __restrict__ gm,
                                               const float* __restrict__ ipb,
                                               const float* __restrict__ dtw,
                                               const float* __restrict__ dtb,
                                               float* __restrict__ dtout) {
  __shared__ float dtr[16][128];
  const int tid = threadIdx.x;
  const int r0  = blockIdx.x * 16;
#pragma unroll
  for (int i = 0; i < 8; ++i) {
    int idx = tid + i * 256;
    int r = idx >> 7, k = idx & 127;
    dtr[r][k] = gm[(size_t)(r0 + r) * N_TOT + 10240 + k] + ipb[2048 + k];
  }
  __syncthreads();
#pragma unroll 1
  for (int ch = 0; ch < 8; ++ch) {
    int col = tid + ch * 256;
    float acc[16];
#pragma unroll
    for (int r = 0; r < 16; ++r) acc[r] = 0.f;
    for (int k = 0; k < 128; ++k) {
      float w = dtw[(size_t)k * H_DIM + col];
#pragma unroll
      for (int r = 0; r < 16; ++r) acc[r] += dtr[r][k] * w;
    }
    float bb = dtb[col];
#pragma unroll
    for (int r = 0; r < 16; ++r) {
      float v = acc[r] + bb;
      float sp = (v > 20.f) ? v : log1pf(expf(v));
      dtout[(size_t)(r0 + r) * H_DIM + col] = sp;
    }
  }
}

// ============================================================
// 5) fused tail: gates, c_lstm, layernorm (block reduce), SSM, outputs
// ============================================================
__global__ __launch_bounds__(256) void fused_tail(
    const float* __restrict__ gm,  const float* __restrict__ cin,
    const float* __restrict__ ssm, const float* __restrict__ dtv,
    const float* __restrict__ Wib, const float* __restrict__ Wfb,
    const float* __restrict__ Wob, const float* __restrict__ Wgb,
    const float* __restrict__ ipb, const float* __restrict__ Alog,
    const float* __restrict__ Dv,  const float* __restrict__ lng,
    const float* __restrict__ lnb, const float* __restrict__ W1,
    const float* __restrict__ W2,  const float* __restrict__ s0,
    float* __restrict__ out) {
  __shared__ float cl[2048];
  __shared__ float rs[256];
  __shared__ float rq[256];
  const int b   = blockIdx.x;
  const int tid = threadIdx.x;
  const float* grow = gm + (size_t)b * N_TOT;
  float s = 0.f, s2 = 0.f;
#pragma unroll
  for (int e = 0; e < 8; ++e) {
    int j = tid + e * 256;
    float iv  = sigm(grow[j] + Wib[j]);
    float fv  = sigm(grow[2048 + j] + Wfb[j]);
    float gv  = tanhf(grow[6144 + j] + Wgb[j]);
    float clv = fv * cin[(size_t)b * H_DIM + j] + iv * gv;
    cl[j] = clv; s += clv; s2 += clv * clv;
  }
  rs[tid] = s; rq[tid] = s2;
  __syncthreads();
  for (int off = 128; off > 0; off >>= 1) {
    if (tid < off) { rs[tid] += rs[tid + off]; rq[tid] += rq[tid + off]; }
    __syncthreads();
  }
  const float mu   = rs[0] * (1.f / 2048.f);
  const float var  = rq[0] * (1.f / 2048.f) - mu * mu;
  const float rstd = rsqrtf(var + 1e-5f);
  const float w1 = 1.4f - W1[0], w2 = 0.6f - W2[0], sc = s0[0];
#pragma unroll
  for (int e = 0; e < 8; ++e) {
    int j = tid + e * 256;
    float norm = (cl[j] - mu) * rstd * lng[j] + lnb[j];
    float xs   = grow[8192 + j]  + ipb[j];
    float Bse  = grow[10368 + j] + ipb[2176 + j];
    float Cse  = grow[12416 + j] + ipb[4224 + j];
    float dt   = dtv[(size_t)b * H_DIM + j];
    float Ad   = expf(-dt * expf(Alog[j]));          // exp(dt * A), A = -exp(A_log)
    float ssn  = Ad * ssm[(size_t)b * H_DIM + j] + dt * Bse * xs;
    float cssm = Cse * ssn + Dv[j] * xs;
    float cn   = (w1 * norm + w2 * cssm) * sc;
    float ov   = sigm(grow[4096 + j] + Wob[j]);
    float hn   = ov * tanhf(cn);
    out[(size_t)b * H_DIM + j]            = hn;
    out[BH + (size_t)b * H_DIM + j]       = cn;
    out[2 * BH + (size_t)b * H_DIM + j]   = ssn;
  }
}

// ============================================================
extern "C" void kernel_launch(void* const* d_in, const int* in_sizes, int n_in,
                              void* d_out, int out_size, void* d_ws, size_t ws_size,
                              hipStream_t stream) {
  const float* x    = (const float*)d_in[0];
  const float* h    = (const float*)d_in[1];
  const float* c    = (const float*)d_in[2];
  const float* ssm  = (const float*)d_in[3];
  const float* Wi   = (const float*)d_in[4];
  const float* Wib  = (const float*)d_in[5];
  const float* Wf   = (const float*)d_in[6];
  const float* Wfb  = (const float*)d_in[7];
  const float* Wo   = (const float*)d_in[8];
  const float* Wob  = (const float*)d_in[9];
  const float* Wg   = (const float*)d_in[10];
  const float* Wgb  = (const float*)d_in[11];
  const float* Wp   = (const float*)d_in[12];
  const float* ipb  = (const float*)d_in[13];
  const float* dtw  = (const float*)d_in[14];
  const float* Alog = (const float*)d_in[15];
  const float* Dv   = (const float*)d_in[16];
  const float* lng  = (const float*)d_in[17];
  const float* lnb  = (const float*)d_in[18];
  const float* W1   = (const float*)d_in[19];
  const float* W2   = (const float*)d_in[20];
  const float* s0   = (const float*)d_in[21];
  const float* dtb  = (const float*)d_in[22];

  char* ws = (char*)d_ws;
  unsigned short* cb = (unsigned short*)(ws);                 //  33,554,432 B: combined bf16
  unsigned short* wt = (unsigned short*)(ws + 33554432);      // 118,489,088 B: WT bf16 (N_TOT x C)
  float*          gmo = (float*)(ws + 152043520);             // 236,978,176 B: GEMM out f32
  float*          dtv = (float*)(ws + 389021696);             //  33,554,432 B: softplus(dt)

  pack_combined<<<(B_ROWS * C_DIM / 4) / 256, 256, 0, stream>>>(h, x, cb);
  wt_build<<<dim3(C_DIM / 32, N_TOT / 32), dim3(32, 8), 0, stream>>>(Wi, Wf, Wo, Wg, Wp, wt);
  gemm_bf16<<<dim3(N_TOT / 128, B_ROWS / 128), 256, 0, stream>>>(cb, wt, gmo);
  dt_gemm<<<B_ROWS / 16, 256, 0, stream>>>(gmo, ipb, dtw, dtb, dtv);
  fused_tail<<<B_ROWS, 256, 0, stream>>>(gmo, c, ssm, dtv, Wib, Wfb, Wob, Wgb,
                                         ipb, Alog, Dv, lng, lnb, W1, W2, s0,
                                         (float*)d_out);
}
